// CompressedInteractionNetwork_11072425689398
// MI455X (gfx1250) — matrix operand order repaired and checked by reading, not compiled, not run on hardware
//
#include <hip/hip_runtime.h>

typedef __attribute__((ext_vector_type(16))) __bf16 v16bf;
typedef __attribute__((ext_vector_type(2)))  __bf16 bf16x2;
typedef __attribute__((ext_vector_type(8)))  float  v8f;

#define NF      32
#define ED      16
#define NBATCH  4096
#define CS      128
#define KCHUNK  512
#define ZROW    (KCHUNK + 8)   // pad 8 halves (16B) -> conflict-free b128 B-frag loads
#define YROW    (CS + 4)       // pad 4 floats -> spread banks on y-tile stores

__device__ __forceinline__ unsigned short f2bf(float f) {
    unsigned int u = __float_as_uint(f);
    u += 0x7FFFu + ((u >> 16) & 1u);      // round-to-nearest-even
    return (unsigned short)(u >> 16);
}

// native bf16 pair conversion: lets the backend use packed f32->bf16 cvt
__device__ __forceinline__ unsigned int pack2bf(float a, float b) {
    union { bf16x2 v; unsigned int u; } cv;
    cv.v.x = (__bf16)a;
    cv.v.y = (__bf16)b;
    return cv.u;
}

__global__ void cvt_f32_bf16(const float* __restrict__ src,
                             unsigned short* __restrict__ dst, int n) {
    int i = blockIdx.x * blockDim.x + threadIdx.x;
    if (i < n) dst[i] = f2bf(src[i]);
}

__global__ __launch_bounds__(256)
void cin_wmma_kernel(const float* __restrict__ x,
                     const unsigned short* __restrict__ W0,
                     const unsigned short* __restrict__ W1,
                     const unsigned short* __restrict__ W2,
                     const float* __restrict__ b0,
                     const float* __restrict__ b1,
                     const float* __restrict__ b2,
                     float* __restrict__ out) {
    // LDS: z chunk (bf16, e-major, padded), x (e-major), double y buffer (e-major, padded)
    __shared__ __align__(16) unsigned short zT[16 * ZROW];          // 16.6 KB
    __shared__ __align__(16) float xsT[16 * NF];                    //  2.0 KB
    __shared__ __align__(16) float ybT[2][16 * YROW];               // 16.9 KB

    const int tid    = threadIdx.x;
    const int lane   = tid & 31;
    const int wave   = tid >> 5;        // 0..7 == M-tile index
    const int e_lane = lane & 15;       // N index (embed dim) for B/C/D fragments
    const int grp    = lane >> 4;       // lane half-group
    const int b      = blockIdx.x;      // batch element

    // stage x[b] (32x16 f32) into LDS transposed: xsT[e][f]
    const float* xb = x + (size_t)b * (NF * ED);
    for (int i = tid; i < NF * ED; i += 256) {
        int f = i >> 4, e = i & 15;
        xsT[e * NF + f] = xb[i];
    }
    __syncthreads();

    const unsigned short* Ws[3] = {W0, W1, W2};
    const float*          bs[3] = {b0, b1, b2};

    float* outb    = out + (size_t)b * 256;
    int    out_off = 0;

    unsigned int* zT32 = (unsigned int*)zT;

    #pragma unroll
    for (int layer = 0; layer < 3; ++layer) {
        const int G    = (layer == 0) ? 32 : 64;
        const int logG = (layer == 0) ? 5 : 6;
        const int C    = NF * G;                     // 1024 or 2048

        // h source: layer 0 reads x itself; later layers read prev y[64:128]
        const float* hptr    = (layer == 0) ? xsT : &ybT[(layer + 1) & 1][64];
        const int    hstride = (layer == 0) ? NF : YROW;   // compile-time after unroll
        float* yw = ybT[layer & 1];

        // accumulator init = bias (broadcast loads, L2-resident)
        v8f acc;
        {
            const float* bl    = bs[layer];
            const int    obase = wave * 16 + grp * 8;
            #pragma unroll
            for (int v = 0; v < 8; ++v) acc[v] = bl[obase + v];
        }

        // A-matrix row for this lane: W row (wave*16 + M), M = lane%16
        const unsigned short* wrow = Ws[layer] + (size_t)(wave * 16 + e_lane) * C;

        for (int k0 = 0; k0 < C; k0 += KCHUNK) {
            // ---- pair-packed cooperative Z build ----
            // thread owns kc = 2*tid (f,g fixed for the whole chunk), loops e
            {
                const int c = k0 + tid * 2;
                const int f = c >> logG;
                const int g = c & (G - 1);            // even -> float2-aligned
                const float*  xcol = xsT + f;
                const float*  hcol = hptr + g;
                unsigned int* zcol = zT32 + tid;
                #pragma unroll
                for (int e = 0; e < 16; ++e) {
                    float  xv = xcol[e * NF];
                    float2 hv = *(const float2*)(hcol + e * hstride);
                    zcol[e * (ZROW / 2)] = pack2bf(xv * hv.x, xv * hv.y);
                }
            }
            __syncthreads();

            // hint next W chunk toward the caches
            if (k0 + KCHUNK < C)
                __builtin_prefetch(wrow + k0 + KCHUNK, 0, 0);

            // ---- WMMA over the chunk: 16 K-tiles of 32 ----
            #pragma unroll 4
            for (int kt = 0; kt < KCHUNK; kt += 32) {
                union { v16bf v; uint4 q[2]; } A, B;
                // A 16x32 bf16: elems 0-7 -> K = grp*8..+7 ; elems 8-15 -> K = 16+grp*8..+7
                const unsigned short* ap = wrow + (k0 + kt + grp * 8);
                A.q[0] = *(const uint4*)(ap);
                A.q[1] = *(const uint4*)(ap + 16);
                // B 32x16 bf16: elems 0-7 -> K = grp*8..+7 ; elems 8-15 -> K = 16+grp*8..+7 at fixed e
                const unsigned short* bp = &zT[e_lane * ZROW + kt + grp * 8];
                B.q[0] = *(const uint4*)(bp);
                B.q[1] = *(const uint4*)(bp + 16);
                acc = __builtin_amdgcn_wmma_f32_16x16x32_bf16(
                          false, A.v, false, B.v, (short)0, acc, false, false);
            }
            __syncthreads();
        }

        // ---- ReLU + park y tile in LDS (e-major): yw[e][o] ----
        {
            const int obase = wave * 16 + grp * 8;
            #pragma unroll
            for (int v = 0; v < 8; ++v) {
                float r = acc[v] > 0.0f ? acc[v] : 0.0f;
                yw[e_lane * YROW + (obase + v)] = r;
            }
        }
        __syncthreads();

        // ---- output contribution: sum over e of kept channels ----
        const int nkeep = (layer == 2) ? 128 : 64;
        if (tid < nkeep) {
            float s = 0.0f;
            #pragma unroll
            for (int e = 0; e < 16; ++e) s += yw[e * YROW + tid];
            outb[out_off + tid] = s;
        }
        out_off += nkeep;
        __syncthreads();
    }
}

extern "C" void kernel_launch(void* const* d_in, const int* in_sizes, int n_in,
                              void* d_out, int out_size, void* d_ws, size_t ws_size,
                              hipStream_t stream) {
    (void)in_sizes; (void)n_in; (void)out_size; (void)ws_size;
    const float* x  = (const float*)d_in[0];
    const float* W0 = (const float*)d_in[1];
    const float* b0 = (const float*)d_in[2];
    const float* W1 = (const float*)d_in[3];
    const float* b1 = (const float*)d_in[4];
    const float* W2 = (const float*)d_in[5];
    const float* b2 = (const float*)d_in[6];

    const int n0 = CS * (NF * NF);        // 128*1024
    const int n1 = CS * (NF * 64);        // 128*2048
    const int n2 = CS * (NF * 64);        // 128*2048

    unsigned short* W0b = (unsigned short*)d_ws;
    unsigned short* W1b = W0b + n0;
    unsigned short* W2b = W1b + n1;

    // prologue: W f32 -> bf16 into workspace (L2-resident, reused 4096x)
    cvt_f32_bf16<<<(n0 + 255) / 256, 256, 0, stream>>>(W0, W0b, n0);
    cvt_f32_bf16<<<(n1 + 255) / 256, 256, 0, stream>>>(W1, W1b, n1);
    cvt_f32_bf16<<<(n2 + 255) / 256, 256, 0, stream>>>(W2, W2b, n2);

    cin_wmma_kernel<<<NBATCH, 256, 0, stream>>>(x, W0b, W1b, W2b, b0, b1, b2,
                                                (float*)d_out);
}
